// PointConvDensitySetAbstraction_83339545411693
// MI455X (gfx1250) — matrix operand order repaired and checked by reading, not compiled, not run on hardware
//
#include <hip/hip_runtime.h>
#include <math.h>

// PointConv Density Set Abstraction for MI455X (gfx1250, wave32, WMMA).
// B=8, N=4096, S=1024 (NPOINT), K=32 (NSAMPLE).
// Dense layers run on v_wmma_f32_16x16x32_f16 (f16 operands, f32 accumulate).
// Inter-layer activations are kept in f16 (they are only ever WMMA A-operands),
// halving HBM traffic; BN(train) stats are fused into GEMM epilogues.
// Fragment assembly uses 16B LDS vector loads (ds_load_b128) per the ISA layouts.
// f16 X tiles are staged with GLOBAL_LOAD_ASYNC_TO_LDS_B128 (ASYNCcnt).

#define Bb   8
#define Nn   4096
#define Ss   1024
#define Kk   32
#define BSK  (Bb*Ss*Kk)   // 262144
#define BNr  (Bb*Nn)      // 32768
#define CDIV(a,b) (((a)+(b)-1)/(b))

typedef _Float16 v16h __attribute__((ext_vector_type(16)));
typedef _Float16 v8h  __attribute__((ext_vector_type(8)));
typedef float    v8f  __attribute__((ext_vector_type(8)));
typedef int      v4i  __attribute__((vector_size(16)));   // builtin's pointee type

union PkH2 { _Float16 h[2]; unsigned int u; };

#if defined(__has_builtin)
#if __has_builtin(__builtin_amdgcn_global_load_async_to_lds_b128)
#define HAVE_ASYNC_LDS 1
#endif
#endif

#ifdef HAVE_ASYNC_LDS
#define AS1 __attribute__((address_space(1)))
#define AS3 __attribute__((address_space(3)))
__device__ __forceinline__ void async_wait0() {
#if __has_builtin(__builtin_amdgcn_s_wait_asynccnt)
  __builtin_amdgcn_s_wait_asynccnt(0);
#else
  asm volatile("s_wait_asynccnt 0x0" ::: "memory");
#endif
}
#endif

// ---------------------------------------------------------------- transpose in
__global__ void k_transpose_in(const float* __restrict__ xyz,
                               const float* __restrict__ pts,
                               float* __restrict__ xyzT,
                               float* __restrict__ ptsT) {
  int idx = blockIdx.x * 256 + threadIdx.x;   // over B*N
  if (idx >= Bb * Nn) return;
  int b = idx >> 12, n = idx & (Nn - 1);
  #pragma unroll
  for (int c = 0; c < 3; ++c) {
    xyzT[(size_t)idx * 3 + c] = xyz[((size_t)b * 3 + c) * Nn + n];
    ptsT[(size_t)idx * 3 + c] = pts[((size_t)b * 3 + c) * Nn + n];
  }
}

// ---------------------------------------------------------------- density (N^2)
__global__ __launch_bounds__(128) void k_density(const float* __restrict__ xyzT,
                                                 float* __restrict__ dens) {
  const int b  = blockIdx.x >> 5;             // N/128 = 32 chunks per batch
  const int i0 = (blockIdx.x & 31) * 128;
  const int tid = threadIdx.x;
  __shared__ float sx[128][3];
  const float* base = xyzT + (size_t)b * Nn * 3;
  int i = i0 + tid;
  float px = base[i*3+0], py = base[i*3+1], pz = base[i*3+2];
  float acc = 0.f;
  for (int j0 = 0; j0 < Nn; j0 += 128) {
    __syncthreads();
    sx[tid][0] = base[(j0+tid)*3+0];
    sx[tid][1] = base[(j0+tid)*3+1];
    sx[tid][2] = base[(j0+tid)*3+2];
    __syncthreads();
    #pragma unroll 4
    for (int j = 0; j < 128; ++j) {
      float dx = px - sx[j][0], dy = py - sx[j][1], dz = pz - sx[j][2];
      float d = dx*dx + dy*dy + dz*dz;
      acc += __expf(-50.f * d);               // exp(-d/(2*0.1^2))
    }
  }
  dens[(size_t)b*Nn + i] = acc * (4.0f / (float)Nn);  // /(2.5*0.1) then mean
}

// ---------------------------------------------------------------- small dense layer (+BN stats)
__global__ __launch_bounds__(256) void k_small_layer(const float* __restrict__ X,
                                                     const float* __restrict__ W,
                                                     const float* __restrict__ bias,
                                                     float* __restrict__ Y,
                                                     int M, int Cin, int Cout,
                                                     float* __restrict__ stats) {
  __shared__ float ss[16], sq[16];
  if (threadIdx.x < 16) { ss[threadIdx.x] = 0.f; sq[threadIdx.x] = 0.f; }
  __syncthreads();
  int row = blockIdx.x * 256 + threadIdx.x;
  if (row < M) {
    float xin[8];
    for (int c = 0; c < Cin; ++c) xin[c] = X[(size_t)row * Cin + c];
    for (int o = 0; o < Cout; ++o) {
      float a = bias[o];
      for (int c = 0; c < Cin; ++c) a += W[o * Cin + c] * xin[c];
      Y[(size_t)row * Cout + o] = a;
      atomicAdd(&ss[o], a);
      atomicAdd(&sq[o], a * a);
    }
  }
  __syncthreads();
  if (threadIdx.x < Cout) {
    atomicAdd(&stats[threadIdx.x],        ss[threadIdx.x]);
    atomicAdd(&stats[Cout + threadIdx.x], sq[threadIdx.x]);
  }
}

// ---------------------------------------------------------------- BN(train)+ReLU, f32 in-place
__global__ void k_bn_apply(float* __restrict__ Y, const float* __restrict__ stats,
                           const float* __restrict__ g, const float* __restrict__ be,
                           int Nch, float invM, int total) {
  int idx = blockIdx.x * 256 + threadIdx.x;
  if (idx >= total) return;
  int n = idx % Nch;
  float mean = stats[n] * invM;
  float var  = stats[Nch + n] * invM - mean * mean;
  float sc   = g[n] * rsqrtf(var + 1e-5f);
  float v    = (Y[idx] - mean) * sc + be[n];
  Y[idx] = fmaxf(v, 0.f);
}

// ---------------------------------------------------------------- BN(train)+ReLU, f32 -> f16 out
__global__ void k_bn_apply_h(const float* __restrict__ Y, _Float16* __restrict__ O,
                             const float* __restrict__ stats,
                             const float* __restrict__ g, const float* __restrict__ be,
                             int Nch, float invM, int total) {
  int idx = blockIdx.x * 256 + threadIdx.x;
  if (idx >= total) return;
  int n = idx % Nch;
  float mean = stats[n] * invM;
  float var  = stats[Nch + n] * invM - mean * mean;
  float sc   = g[n] * rsqrtf(var + 1e-5f);
  float v    = (Y[idx] - mean) * sc + be[n];
  O[idx] = (_Float16)fmaxf(v, 0.f);
}

__global__ void k_zero(float* p, int n) {
  int i = blockIdx.x * 256 + threadIdx.x;
  if (i < n) p[i] = 0.f;
}

// ---------------------------------------------------------------- FPS (serial per batch)
__global__ __launch_bounds__(256) void k_fps(const float* __restrict__ xyzT,
                                             int* __restrict__ fps_idx,
                                             float* __restrict__ new_xyz) {
  const int b = blockIdx.x;
  const int tid = threadIdx.x;
  __shared__ float dist[Nn];
  __shared__ float rv[256];
  __shared__ int   ri[256];
  __shared__ int   sfar;
  __shared__ float c3[3];
  const float* base = xyzT + (size_t)b * Nn * 3;
  for (int j = tid; j < Nn; j += 256) dist[j] = 1e10f;
  if (tid == 0) sfar = 0;
  __syncthreads();
  for (int i = 0; i < Ss; ++i) {
    int far = sfar;
    if (tid == 0) fps_idx[b * Ss + i] = far;
    if (tid < 3)  c3[tid] = base[far * 3 + tid];
    __syncthreads();
    float cx = c3[0], cy = c3[1], cz = c3[2];
    if (tid == 0) {
      new_xyz[((size_t)b*Ss + i)*3 + 0] = cx;
      new_xyz[((size_t)b*Ss + i)*3 + 1] = cy;
      new_xyz[((size_t)b*Ss + i)*3 + 2] = cz;
    }
    float lmax = -1.f; int lidx = 0;
    for (int j = tid; j < Nn; j += 256) {
      float dx = base[j*3+0]-cx, dy = base[j*3+1]-cy, dz = base[j*3+2]-cz;
      float d = dx*dx + dy*dy + dz*dz;
      float dm = fminf(dist[j], d);
      dist[j] = dm;
      if (dm > lmax) { lmax = dm; lidx = j; }
    }
    rv[tid] = lmax; ri[tid] = lidx;
    __syncthreads();
    for (int s = 128; s > 0; s >>= 1) {
      if (tid < s) {
        if (rv[tid+s] > rv[tid] || (rv[tid+s] == rv[tid] && ri[tid+s] < ri[tid])) {
          rv[tid] = rv[tid+s]; ri[tid] = ri[tid+s];
        }
      }
      __syncthreads();
    }
    if (tid == 0) sfar = ri[0];
    __syncthreads();
  }
}

// ---------------------------------------------------------------- KNN(32) + gather
__global__ __launch_bounds__(128) void k_knn_gather(const float* __restrict__ xyzT,
                                                    const float* __restrict__ ptsT,
                                                    const float* __restrict__ nxyz,
                                                    const float* __restrict__ dscale,
                                                    float* __restrict__ npin,   // [BS,K,6]
                                                    float* __restrict__ gx) {   // [BS,K,3]
  const int bs = blockIdx.x;
  const int b  = bs >> 10;
  const int tid = threadIdx.x;
  __shared__ float dist[Nn];
  __shared__ float rv[128];
  __shared__ int   ri[128];
  __shared__ int   sel[Kk];
  const float* base = xyzT + (size_t)b * Nn * 3;
  float nx = nxyz[(size_t)bs*3+0], ny = nxyz[(size_t)bs*3+1], nz = nxyz[(size_t)bs*3+2];
  for (int j = tid; j < Nn; j += 128) {
    float dx = base[j*3+0]-nx, dy = base[j*3+1]-ny, dz = base[j*3+2]-nz;
    dist[j] = dx*dx + dy*dy + dz*dz;
  }
  __syncthreads();
  for (int k = 0; k < Kk; ++k) {
    float lv = 3.0e38f; int li = 0;
    for (int j = tid; j < Nn; j += 128) {
      float d = dist[j];
      if (d < lv) { lv = d; li = j; }
    }
    rv[tid] = lv; ri[tid] = li;
    __syncthreads();
    for (int s = 64; s > 0; s >>= 1) {
      if (tid < s) {
        if (rv[tid+s] < rv[tid] || (rv[tid+s] == rv[tid] && ri[tid+s] < ri[tid])) {
          rv[tid] = rv[tid+s]; ri[tid] = ri[tid+s];
        }
      }
      __syncthreads();
    }
    if (tid == 0) { sel[k] = ri[0]; dist[ri[0]] = 3.4e38f; }
    __syncthreads();
  }
  if (tid < Kk) {
    int j = sel[tid];
    float ds = dscale[(size_t)b*Nn + j];
    size_t ob = (size_t)bs * Kk + tid;
    float cc[3] = {nx, ny, nz};
    #pragma unroll
    for (int c = 0; c < 3; ++c) {
      float nv = base[j*3 + c] - cc[c];
      npin[ob*6 + c]     = nv;
      npin[ob*6 + 3 + c] = ptsT[((size_t)b*Nn + j)*3 + c];
      gx[ob*3 + c]       = nv * ds;
    }
  }
}

// ---------------------------------------------------------------- WMMA GEMM: Y = X*W^T + b (+BN stats)
// XH=true: X is _Float16 [M,ldx] (M % 128 == 0 required); XH=false: X is f32.
// W: [NT*16, Kd] f32 row-major. Y: [M, NT*16] f32 pre-activation.
template <int NT, bool XH>
__global__ __launch_bounds__(256) void k_wmma_gemm(const void* __restrict__ Xv,
                                                   const float* __restrict__ W,
                                                   const float* __restrict__ bias,
                                                   float* __restrict__ Y,
                                                   int M, int Kd, int ldx,
                                                   float* __restrict__ stats) {
  __shared__ __align__(16) _Float16 sX[128][32];
  __shared__ __align__(16) _Float16 sW[NT * 16][32];
  __shared__ float sSum[NT * 16], sSq[NT * 16];
  const int tid  = threadIdx.x;
  const int lane = tid & 31, wv = tid >> 5;
  const int m0 = blockIdx.x * 128;
  if (tid < NT * 16) { sSum[tid] = 0.f; sSq[tid] = 0.f; }

  v8f zero = {};
  v8f acc[NT];
  #pragma unroll
  for (int t = 0; t < NT; ++t) acc[t] = zero;

  for (int k0 = 0; k0 < Kd; k0 += 32) {
    __syncthreads();
    if (XH) {
      const _Float16* X = (const _Float16*)Xv;
#ifdef HAVE_ASYNC_LDS
      // per-lane 16B async copies global -> LDS (ASYNCcnt), no VGPR staging
      #pragma unroll
      for (int i = 0; i < 2; ++i) {
        int e = (i * 256 + tid) * 8;
        int r = e >> 5, c = e & 31;
        const _Float16* gp = &X[(size_t)(m0 + r) * ldx + (k0 + c)];
        __builtin_amdgcn_global_load_async_to_lds_b128(
            (AS1 v4i*)gp, (AS3 v4i*)&sX[r][c], 0, 0);
      }
#else
      #pragma unroll
      for (int i = 0; i < 2; ++i) {
        int e = (i * 256 + tid) * 8;
        int r = e >> 5, c = e & 31;
        int gr = m0 + r;
        v8h v = {};
        if (gr < M) v = *(const v8h*)&X[(size_t)gr * ldx + (k0 + c)];
        *(v8h*)&sX[r][c] = v;
      }
#endif
      // prefetch next K-chunk of X
      if (k0 + 32 < Kd) {
        int e = tid * 8;
        int r = e >> 5, c = e & 31;
        __builtin_prefetch(&X[(size_t)(m0 + r) * ldx + (k0 + 32 + c)], 0, 1);
      }
    } else {
      const float* X = (const float*)Xv;
      #pragma unroll
      for (int i = 0; i < 8; ++i) {             // pairs -> packed b32 LDS store
        int e = (i * 256 + tid) * 2;
        int r = e >> 5, c = e & 31;
        int gr = m0 + r;
        PkH2 pk;
        pk.h[0] = (_Float16)((gr < M && (k0 + c)     < Kd) ? X[(size_t)gr * ldx + k0 + c]     : 0.f);
        pk.h[1] = (_Float16)((gr < M && (k0 + c + 1) < Kd) ? X[(size_t)gr * ldx + k0 + c + 1] : 0.f);
        *(unsigned int*)&sX[r][c] = pk.u;
      }
    }
    #pragma unroll
    for (int i = 0; i < NT; ++i) {              // (NT*16)x32 W tile, packed pairs
      int e = (i * 256 + tid) * 2;
      int r = e >> 5, c = e & 31;
      PkH2 pk;
      pk.h[0] = (_Float16)(((k0 + c)     < Kd) ? W[(size_t)r * Kd + k0 + c]     : 0.f);
      pk.h[1] = (_Float16)(((k0 + c + 1) < Kd) ? W[(size_t)r * Kd + k0 + c + 1] : 0.f);
      *(unsigned int*)&sW[r][c] = pk.u;
    }
#ifdef HAVE_ASYNC_LDS
    if (XH) async_wait0();                      // own async transfers done
#endif
    __syncthreads();                            // all waves' tiles visible
    // A fragment: m = lane&15 (+wave tile); K = {kg..kg+7, 16+kg..16+kg+7}, kg = (lane>>4)*8
    v16h a;
    {
      int m  = (lane & 15) + wv * 16;
      int kg = (lane >> 4) * 8;
      v8h lo = *(const v8h*)&sX[m][kg];
      v8h hi = *(const v8h*)&sX[m][16 + kg];
      a = __builtin_shufflevector(lo, hi, 0,1,2,3,4,5,6,7,8,9,10,11,12,13,14,15);
    }
    #pragma unroll
    for (int t = 0; t < NT; ++t) {
      // B fragment: n = lane&15; K = {kb..kb+7, 16+kb..16+kb+7}, kb = (lane>>4)*8
      int n  = (lane & 15) + t * 16;
      int kb = (lane >> 4) * 8;
      v8h b0 = *(const v8h*)&sW[n][kb];
      v8h b1 = *(const v8h*)&sW[n][16 + kb];
      v16h bfrag = __builtin_shufflevector(b0, b1, 0,1,2,3,4,5,6,7,8,9,10,11,12,13,14,15);
      acc[t] = __builtin_amdgcn_wmma_f32_16x16x32_f16(
          false, a, false, bfrag, (short)0, acc[t], false, false);
    }
  }

  // epilogue: bias add, store pre-BN, per-channel sum/sumsq
  #pragma unroll
  for (int t = 0; t < NT; ++t) {
    int n = t * 16 + (lane & 15);
    float bv = bias ? bias[n] : 0.f;
    float ls = 0.f, lq = 0.f;
    #pragma unroll
    for (int r = 0; r < 8; ++r) {
      int mrow = m0 + wv * 16 + r + ((lane >> 4) << 3);
      float v = acc[t][r] + bv;
      if (mrow < M) Y[(size_t)mrow * (NT * 16) + n] = v;
      ls += v; lq += v * v;
    }
    if (stats) { atomicAdd(&sSum[n], ls); atomicAdd(&sSq[n], lq); }
  }
  if (stats) {
    __syncthreads();
    if (tid < NT * 16) {
      atomicAdd(&stats[tid],            sSum[tid]);
      atomicAdd(&stats[NT * 16 + tid],  sSq[tid]);
    }
  }
}

// ---------------------------------------------------------------- per-(b,s) einsum: agg[c,w] = sum_k np[k,c]*wt[k,w]
// NP: [BS,32,128] f16, WT: [BS,32,16] f16, AGG: [BS,2048] f16.
__global__ __launch_bounds__(256) void k_agg(const _Float16* __restrict__ NP,
                                             const _Float16* __restrict__ WT,
                                             _Float16* __restrict__ AGG) {
  // K-major LDS tiles (transposed on write) so fragment reads are b128 loads.
  // Row pad of 8 halves keeps 16B alignment and staggers banks.
  __shared__ __align__(16) _Float16 sNPt[128][40];  // [c][k]
  __shared__ __align__(16) _Float16 sWTt[16][40];   // [w][k]
  const int tid = threadIdx.x;
  const size_t bs = blockIdx.x;
  const _Float16* np = NP + bs * (32 * 128);
  const _Float16* wt = WT + bs * (32 * 16);
  #pragma unroll
  for (int i = 0; i < 2; ++i) {
    int e = (i * 256 + tid) * 8;                 // halves; c-contiguous in global
    v8h v = *(const v8h*)&np[e];
    int k = e >> 7, c = e & 127;
    #pragma unroll
    for (int j = 0; j < 8; ++j) sNPt[c + j][k] = v[j];
  }
  for (int i = tid; i < 512; i += 256) {
    int k = i >> 4, w = i & 15;
    sWTt[w][k] = wt[i];
  }
  __syncthreads();
  const int lane = tid & 31, wv = tid >> 5;      // wave wv owns c-tile wv (8 tiles of 16)
  v16h a, bfrag;
  {
    int c  = wv * 16 + (lane & 15);
    int kg = (lane >> 4) * 8;
    v8h lo = *(const v8h*)&sNPt[c][kg];
    v8h hi = *(const v8h*)&sNPt[c][16 + kg];
    a = __builtin_shufflevector(lo, hi, 0,1,2,3,4,5,6,7,8,9,10,11,12,13,14,15);
    int w  = lane & 15;
    int kb = (lane >> 4) * 8;
    v8h b0 = *(const v8h*)&sWTt[w][kb];
    v8h b1 = *(const v8h*)&sWTt[w][16 + kb];
    bfrag = __builtin_shufflevector(b0, b1, 0,1,2,3,4,5,6,7,8,9,10,11,12,13,14,15);
  }
  v8f d = {};
  d = __builtin_amdgcn_wmma_f32_16x16x32_f16(false, a, false, bfrag, (short)0, d, false, false);
  _Float16* out = AGG + bs * 2048;
  #pragma unroll
  for (int r = 0; r < 8; ++r) {
    int c = wv * 16 + r + ((lane >> 4) << 3);
    int w = lane & 15;
    out[c * 16 + w] = (_Float16)d[r];            // C-major flatten: c*16+w
  }
}

// ---------------------------------------------------------------- output assembly (transposes)
__global__ void k_write_out(const float* __restrict__ nxyz,   // [B,S,3]
                            const float* __restrict__ ylin,   // [B,S,128]
                            float* __restrict__ out) {
  int idx = blockIdx.x * 256 + threadIdx.x;
  const int P1 = Bb * 3 * Ss;                 // 24576
  const int P2 = Bb * 128 * Ss;               // 1048576
  if (idx < P1) {
    int b = idx / (3 * Ss); int c = (idx / Ss) % 3; int s = idx % Ss;
    out[idx] = nxyz[((size_t)(b * Ss + s)) * 3 + c];
  } else if (idx < P1 + P2) {
    int j = idx - P1;
    int b = j / (128 * Ss); int o = (j / Ss) % 128; int s = j % Ss;
    out[idx] = ylin[((size_t)(b * Ss + s)) * 128 + o];
  }
}

// ================================================================ host
extern "C" void kernel_launch(void* const* d_in, const int* in_sizes, int n_in,
                              void* d_out, int out_size, void* d_ws, size_t ws_size,
                              hipStream_t stream) {
  (void)in_sizes; (void)n_in; (void)out_size; (void)ws_size;
  const float* xyz = (const float*)d_in[0];
  const float* pts = (const float*)d_in[1];
  // params flattened in insertion order: dn(3x(w,b,g,be)), wn(3x4), mlp(3x4), lin(w,b), bnl(g,be)
  const float *dn_w[3], *dn_b[3], *dn_g[3], *dn_be[3];
  const float *wn_w[3], *wn_b[3], *wn_g[3], *wn_be[3];
  const float *ml_w[3], *ml_b[3], *ml_g[3], *ml_be[3];
  int p = 2;
  for (int l = 0; l < 3; ++l) { dn_w[l]=(const float*)d_in[p++]; dn_b[l]=(const float*)d_in[p++];
                                dn_g[l]=(const float*)d_in[p++]; dn_be[l]=(const float*)d_in[p++]; }
  for (int l = 0; l < 3; ++l) { wn_w[l]=(const float*)d_in[p++]; wn_b[l]=(const float*)d_in[p++];
                                wn_g[l]=(const float*)d_in[p++]; wn_be[l]=(const float*)d_in[p++]; }
  for (int l = 0; l < 3; ++l) { ml_w[l]=(const float*)d_in[p++]; ml_b[l]=(const float*)d_in[p++];
                                ml_g[l]=(const float*)d_in[p++]; ml_be[l]=(const float*)d_in[p++]; }
  const float* lin_w  = (const float*)d_in[p++];
  const float* lin_b  = (const float*)d_in[p++];
  const float* bnl_g  = (const float*)d_in[p++];
  const float* bnl_be = (const float*)d_in[p++];

  float* ws = (float*)d_ws;
  size_t o = 0;
  float* xyzT   = ws + o; o += (size_t)BNr * 3;       // 98304
  float* ptsT   = ws + o; o += (size_t)BNr * 3;       // 98304
  float* dens   = ws + o; o += BNr;                   // 32768
  float* dnA    = ws + o; o += (size_t)BNr * 8;       // 262144
  float* dnB    = ws + o; o += (size_t)BNr * 8;       // 262144
  float* dscale = ws + o; o += BNr;                   // 32768
  int*   fpsidx = (int*)(ws + o); o += Bb * Ss;       // 8192
  float* nxyz   = ws + o; o += (size_t)Bb * Ss * 3;   // 24576
  float* npin   = ws + o; o += (size_t)BSK * 6;       // 1572864
  float* gxb    = ws + o; o += (size_t)BSK * 3;       // 786432
  float* w1b    = ws + o; o += (size_t)BSK * 8;       // in-place BN (feeds small layer)
  float* w2b    = ws + o; o += (size_t)BSK * 8;
  _Float16* h1h = (_Float16*)(ws + o); o += (size_t)BSK * 64 / 2;   // f16 activations
  _Float16* h2h = (_Float16*)(ws + o); o += (size_t)BSK * 64 / 2;
  _Float16* h3h = (_Float16*)(ws + o); o += (size_t)BSK * 128 / 2;
  _Float16* w3h = (_Float16*)(ws + o); o += (size_t)BSK * 16 / 2;
  float* bufP   = ws + o; o += (size_t)BSK * 128;     // shared pre-activation scratch (f32)
  float* stats  = ws + o; o += 256;
  _Float16* aggh = h1h;   // reuse: h1h dead once layer-2 pre-act is computed; agg is 8192*2048 halves

  const float invBN  = 1.0f / (float)BNr;
  const float invBSK = 1.0f / (float)BSK;
  const float invBS  = 1.0f / (float)(Bb * Ss);

  // 1. transposes + density
  k_transpose_in<<<CDIV(BNr,256),256,0,stream>>>(xyz, pts, xyzT, ptsT);
  k_density<<<Bb*(Nn/128),128,0,stream>>>(xyzT, dens);

  // 2. density MLP (1->8->8->1), BN over (B,N)
  k_zero<<<1,256,0,stream>>>(stats, 16);
  k_small_layer<<<CDIV(BNr,256),256,0,stream>>>(dens, dn_w[0], dn_b[0], dnA, BNr, 1, 8, stats);
  k_bn_apply<<<CDIV(BNr*8,256),256,0,stream>>>(dnA, stats, dn_g[0], dn_be[0], 8, invBN, BNr*8);
  k_zero<<<1,256,0,stream>>>(stats, 16);
  k_small_layer<<<CDIV(BNr,256),256,0,stream>>>(dnA, dn_w[1], dn_b[1], dnB, BNr, 8, 8, stats);
  k_bn_apply<<<CDIV(BNr*8,256),256,0,stream>>>(dnB, stats, dn_g[1], dn_be[1], 8, invBN, BNr*8);
  k_zero<<<1,256,0,stream>>>(stats, 2);
  k_small_layer<<<CDIV(BNr,256),256,0,stream>>>(dnB, dn_w[2], dn_b[2], dscale, BNr, 8, 1, stats);
  k_bn_apply<<<CDIV(BNr,256),256,0,stream>>>(dscale, stats, dn_g[2], dn_be[2], 1, invBN, BNr);

  // 3. FPS + KNN + grouping
  k_fps<<<Bb,256,0,stream>>>(xyzT, fpsidx, nxyz);
  k_knn_gather<<<Bb*Ss,128,0,stream>>>(xyzT, ptsT, nxyz, dscale, npin, gxb);

  // 4. weight net (3->8->8->16), BN over (B,S,K); last layer emits f16
  k_zero<<<1,256,0,stream>>>(stats, 16);
  k_small_layer<<<CDIV(BSK,256),256,0,stream>>>(gxb, wn_w[0], wn_b[0], w1b, BSK, 3, 8, stats);
  k_bn_apply<<<CDIV(BSK*8,256),256,0,stream>>>(w1b, stats, wn_g[0], wn_be[0], 8, invBSK, BSK*8);
  k_zero<<<1,256,0,stream>>>(stats, 16);
  k_small_layer<<<CDIV(BSK,256),256,0,stream>>>(w1b, wn_w[1], wn_b[1], w2b, BSK, 8, 8, stats);
  k_bn_apply<<<CDIV(BSK*8,256),256,0,stream>>>(w2b, stats, wn_g[1], wn_be[1], 8, invBSK, BSK*8);
  k_zero<<<1,256,0,stream>>>(stats, 32);
  k_small_layer<<<CDIV(BSK,256),256,0,stream>>>(w2b, wn_w[2], wn_b[2], bufP, BSK, 8, 16, stats);
  k_bn_apply_h<<<CDIV(BSK*16,256),256,0,stream>>>(bufP, w3h, stats, wn_g[2], wn_be[2], 16, invBSK, BSK*16);

  // 5. feature MLP (6->64->64->128) via WMMA, BN over (B,S,K); f16 activations
  k_zero<<<1,256,0,stream>>>(stats, 128);
  k_wmma_gemm<4,false><<<BSK/128,256,0,stream>>>(npin, ml_w[0], ml_b[0], bufP, BSK, 6, 6, stats);
  k_bn_apply_h<<<CDIV(BSK*64,256),256,0,stream>>>(bufP, h1h, stats, ml_g[0], ml_be[0], 64, invBSK, BSK*64);
  k_zero<<<1,256,0,stream>>>(stats, 128);
  k_wmma_gemm<4,true><<<BSK/128,256,0,stream>>>(h1h, ml_w[1], ml_b[1], bufP, BSK, 64, 64, stats);
  k_bn_apply_h<<<CDIV(BSK*64,256),256,0,stream>>>(bufP, h2h, stats, ml_g[1], ml_be[1], 64, invBSK, BSK*64);
  k_zero<<<1,256,0,stream>>>(stats, 256);
  k_wmma_gemm<8,true><<<BSK/128,256,0,stream>>>(h2h, ml_w[2], ml_b[2], bufP, BSK, 64, 64, stats);
  k_bn_apply_h<<<CDIV(BSK*128,256),256,0,stream>>>(bufP, h3h, stats, ml_g[2], ml_be[2], 128, invBSK, BSK*128);

  // 6. per-(b,s) einsum -> agg [B*S, 2048] f16 via WMMA (reuses h1h storage)
  k_agg<<<Bb*Ss,256,0,stream>>>(h3h, w3h, aggh);

  // 7. final linear [8192,2048]x[2048,128] via WMMA + BN over (B,S); result f32 in bufP
  k_zero<<<1,256,0,stream>>>(stats, 256);
  k_wmma_gemm<8,true><<<(Bb*Ss)/128,256,0,stream>>>(aggh, lin_w, lin_b, bufP, Bb*Ss, 2048, 2048, stats);
  k_bn_apply<<<CDIV(Bb*Ss*128,256),256,0,stream>>>(bufP, stats, bnl_g, bnl_be, 128, invBS, Bb*Ss*128);

  // 8. outputs: new_xyz^T [B,3,S] then y^T [B,128,S]
  k_write_out<<<CDIV(Bb*3*Ss + Bb*128*Ss,256),256,0,stream>>>(nxyz, bufP, (float*)d_out);
}
